// BitLinear_8246337208427
// MI455X (gfx1250) — 8-bit matrix operand order repaired and checked by reading, compiled, not run on hardware
//
#include <hip/hip_runtime.h>
#include <hip/hip_bf16.h>

typedef __attribute__((ext_vector_type(8))) int v8i;

#define D_IN   4096
#define D_OUT  4096
#define M_TOT  8192
#define W_ELEMS (D_OUT * D_IN)

#if defined(__HIP_DEVICE_COMPILE__) && \
    __has_builtin(__builtin_amdgcn_global_load_async_to_lds_b128)
#define USE_ASYNC_LDS 1
#endif

typedef int v4i32 __attribute__((vector_size(16)));
typedef __attribute__((address_space(1))) v4i32 as1_v4i;
typedef __attribute__((address_space(3))) v4i32 as3_v4i;

__device__ __forceinline__ void copy_b128_to_lds(const void* gsrc, void* ldst) {
#if defined(USE_ASYNC_LDS)
    __builtin_amdgcn_global_load_async_to_lds_b128(
        (as1_v4i*)(unsigned long long)gsrc,
        (as3_v4i*)(unsigned int)(unsigned long long)ldst,
         0,  0);
#else
    *(int4*)ldst = *(const int4*)gsrc;
#endif
}

template <int N>
__device__ __forceinline__ void async_copy_wait_le() {
#if defined(USE_ASYNC_LDS)
#if __has_builtin(__builtin_amdgcn_s_wait_asynccnt)
    __builtin_amdgcn_s_wait_asynccnt(N);
#else
    asm volatile("s_wait_asynccnt %0" ::"i"(N) : "memory");
#endif
#endif
}

__global__ __launch_bounds__(256) void absum_partial(const float* __restrict__ W,
                                                     float* __restrict__ partial,
                                                     int n4) {
    __shared__ float red[256];
    float s = 0.0f;
    int idx = blockIdx.x * blockDim.x + threadIdx.x;
    int stride = gridDim.x * blockDim.x;
    const float4* W4 = (const float4*)W;
    for (int i = idx; i < n4; i += stride) {
        float4 v = W4[i];
        s += fabsf(v.x) + fabsf(v.y) + fabsf(v.z) + fabsf(v.w);
    }
    red[threadIdx.x] = s;
    __syncthreads();
    for (int o = 128; o > 0; o >>= 1) {
        if (threadIdx.x < o) red[threadIdx.x] += red[threadIdx.x + o];
        __syncthreads();
    }
    if (threadIdx.x == 0) partial[blockIdx.x] = red[0];
}

__global__ __launch_bounds__(256) void absum_final(const float* __restrict__ partial,
                                                   float* __restrict__ scale_out) {
    __shared__ float red[256];
    float s = partial[threadIdx.x] + partial[threadIdx.x + 256] +
              partial[threadIdx.x + 512] + partial[threadIdx.x + 768];
    red[threadIdx.x] = s;
    __syncthreads();
    for (int o = 128; o > 0; o >>= 1) {
        if (threadIdx.x < o) red[threadIdx.x] += red[threadIdx.x + o];
        __syncthreads();
    }
    if (threadIdx.x == 0)
        scale_out[0] = fmaxf(red[0] * (1.0f / (float)W_ELEMS), 1e-6f);
}

__global__ __launch_bounds__(256) void quant_w(const float* __restrict__ W,
                                               const float* __restrict__ scale_p,
                                               unsigned int* __restrict__ qw4,
                                               int n4) {
    const float inv = 1.0f / scale_p[0];
    int idx = blockIdx.x * blockDim.x + threadIdx.x;
    int stride = gridDim.x * blockDim.x;
    const float4* W4 = (const float4*)W;
    for (int i = idx; i < n4; i += stride) {
        float4 v = W4[i];
        int qx = min(1, max(-1, (int)rintf(v.x * inv)));
        int qy = min(1, max(-1, (int)rintf(v.y * inv)));
        int qz = min(1, max(-1, (int)rintf(v.z * inv)));
        int qw = min(1, max(-1, (int)rintf(v.w * inv)));
        qw4[i] = (qx & 0xFF) | ((qy & 0xFF) << 8) | ((qz & 0xFF) << 16) |
                 ((qw & 0xFF) << 24);
    }
}

__global__ __launch_bounds__(256) void quant_x(const float* __restrict__ x,
                                               signed char* __restrict__ qx,
                                               float* __restrict__ row_scale) {
    const int m = blockIdx.x;
    const int tid = threadIdx.x;
    const float4* x4 = (const float4*)(x + (size_t)m * D_IN);

    float4 v[4];
    float mx = 0.0f;
#pragma unroll
    for (int j = 0; j < 4; ++j) {
        v[j] = x4[tid + 256 * j];
        mx = fmaxf(mx, fmaxf(fmaxf(fabsf(v[j].x), fabsf(v[j].y)),
                             fmaxf(fabsf(v[j].z), fabsf(v[j].w))));
    }
    __shared__ float red[256];
    red[tid] = mx;
    __syncthreads();
    for (int o = 128; o > 0; o >>= 1) {
        if (tid < o) red[tid] = fmaxf(red[tid], red[tid + o]);
        __syncthreads();
    }
    const float amax = fmaxf(red[0], 1e-6f);
    const float s = 127.0f / amax;
    if (tid == 0) row_scale[m] = amax * (1.0f / 127.0f);

    unsigned int* q4 = (unsigned int*)(qx + (size_t)m * D_IN);
#pragma unroll
    for (int j = 0; j < 4; ++j) {
        int a = min(127, max(-128, (int)rintf(v[j].x * s)));
        int b = min(127, max(-128, (int)rintf(v[j].y * s)));
        int c = min(127, max(-128, (int)rintf(v[j].z * s)));
        int d = min(127, max(-128, (int)rintf(v[j].w * s)));
        q4[tid + 256 * j] = (a & 0xFF) | ((b & 0xFF) << 8) | ((c & 0xFF) << 16) |
                            ((d & 0xFF) << 24);
    }
}

__global__ __launch_bounds__(256) void bitgemm(const signed char* __restrict__ qx,
                                               const signed char* __restrict__ qw,
                                               const float* __restrict__ row_scale,
                                               const float* __restrict__ scale_p,
                                               const float* __restrict__ bias,
                                               float* __restrict__ out) {
    constexpr int BUF = 128 * 80;
    __shared__ __align__(16) signed char ldsA[3 * BUF];
    __shared__ __align__(16) signed char ldsB[3 * BUF];

    const int tid = threadIdx.x;
    const int lane = tid & 31;
    const int wave = tid >> 5;
    const int wm = wave >> 2;
    const int wn = wave & 3;
    const int m0 = blockIdx.y * 128;
    const int n0 = blockIdx.x * 128;

    const int lrow = lane & 15;
    const int hi = (lane & 16) ? 1 : 0;
    const int akoff = hi ? 8 : 0;
    const int bkoff = hi ? 8 : 0;
    const int crow = hi ? 8 : 0;

    const int aBase = (wm * 64 + lrow) * 80 + akoff;
    const int bBase = (wn * 32 + lrow) * 80 + bkoff;

    const int srow = tid >> 2;
    const int sko = (tid & 3) << 4;
    const signed char* gA0 = qx + (size_t)(m0 + srow) * D_IN + sko;
    const signed char* gA1 = qx + (size_t)(m0 + srow + 64) * D_IN + sko;
    const signed char* gB0 = qw + (size_t)(n0 + srow) * D_IN + sko;
    const signed char* gB1 = qw + (size_t)(n0 + srow + 64) * D_IN + sko;
    const int oL0 = srow * 80 + sko;
    const int oL1 = (srow + 64) * 80 + sko;

    v8i acc[4][2];
#pragma unroll
    for (int t = 0; t < 4; ++t)
#pragma unroll
        for (int u = 0; u < 2; ++u)
#pragma unroll
            for (int r = 0; r < 8; ++r) acc[t][u][r] = 0;

    auto prefetch = [&](int buf, int k0) {
        copy_b128_to_lds(gA0 + k0, ldsA + buf * BUF + oL0);
        copy_b128_to_lds(gA1 + k0, ldsA + buf * BUF + oL1);
        copy_b128_to_lds(gB0 + k0, ldsB + buf * BUF + oL0);
        copy_b128_to_lds(gB1 + k0, ldsB + buf * BUF + oL1);
    };

    const int NK = D_IN / 64;
    prefetch(0, 0);
    prefetch(1, 64);
    async_copy_wait_le<4>();
    __syncthreads();

    int p = 0;
    int q = 2;
#pragma unroll 1
    for (int kt = 0; kt < NK; ++kt) {
        if (kt + 2 < NK) prefetch(q, (kt + 2) * 64);

        const signed char* pAb = ldsA + p * BUF + aBase;
        const signed char* pBb = ldsB + p * BUF + bBase;

        v8i a[4];
#pragma unroll
        for (int t = 0; t < 4; ++t) {
            const signed char* pA = pAb + t * (16 * 80);
            int2 c0 = *(const int2*)(pA + 0);
            int2 c1 = *(const int2*)(pA + 16);
            int2 c2 = *(const int2*)(pA + 32);
            int2 c3 = *(const int2*)(pA + 48);
            a[t][0] = c0.x; a[t][1] = c0.y;
            a[t][2] = c1.x; a[t][3] = c1.y;
            a[t][4] = c2.x; a[t][5] = c2.y;
            a[t][6] = c3.x; a[t][7] = c3.y;
        }
        v8i b[2];
#pragma unroll
        for (int u = 0; u < 2; ++u) {
            const signed char* pB = pBb + u * (16 * 80);
            int2 d0 = *(const int2*)(pB + 0);
            int2 d1 = *(const int2*)(pB + 16);
            int2 d2 = *(const int2*)(pB + 32);
            int2 d3 = *(const int2*)(pB + 48);
            b[u][0] = d0.x; b[u][1] = d0.y; b[u][2] = d1.x; b[u][3] = d1.y;
            b[u][4] = d2.x; b[u][5] = d2.y; b[u][6] = d3.x; b[u][7] = d3.y;
        }
#pragma unroll
        for (int t = 0; t < 4; ++t)
#pragma unroll
            for (int u = 0; u < 2; ++u)
                acc[t][u] = __builtin_amdgcn_wmma_i32_16x16x64_iu8(
                    true, a[t], true, b[u], acc[t][u], false, false);

        if (kt + 2 < NK) {
            async_copy_wait_le<4>();
        } else {
            async_copy_wait_le<0>();
        }
        __syncthreads();

        p = (p == 2) ? 0 : p + 1;
        q = (q == 2) ? 0 : q + 1;
    }

    const float wscale = scale_p[0];
#pragma unroll
    for (int t = 0; t < 4; ++t) {
        const int mbase = m0 + wm * 64 + t * 16 + crow;
        float rs[8];
#pragma unroll
        for (int r = 0; r < 8; ++r) rs[r] = wscale * row_scale[mbase + r];
#pragma unroll
        for (int u = 0; u < 2; ++u) {
            const int n = n0 + wn * 32 + u * 16 + lrow;
            const float bn = bias[n];
#pragma unroll
            for (int r = 0; r < 8; ++r)
                out[(size_t)(mbase + r) * D_OUT + n] =
                    (float)acc[t][u][r] * rs[r] + bn;
        }
    }
}

extern "C" void kernel_launch(void* const* d_in, const int* in_sizes, int n_in,
                              void* d_out, int out_size, void* d_ws, size_t ws_size,
                              hipStream_t stream) {
    const float* x = (const float*)d_in[0];
    const float* W = (const float*)d_in[1];
    const float* b = (const float*)d_in[2];
    float* out = (float*)d_out;

    char* ws = (char*)d_ws;
    float* scale_p   = (float*)(ws + 0);
    float* partial   = (float*)(ws + 256);
    float* row_scale = (float*)(ws + 8192);
    signed char* qx  = (signed char*)(ws + 65536);
    signed char* qw  = (signed char*)(ws + 65536 + (size_t)M_TOT * D_IN);

    absum_partial<<<1024, 256, 0, stream>>>(W, partial, W_ELEMS / 4);
    absum_final<<<1, 256, 0, stream>>>(partial, scale_p);
    quant_w<<<2048, 256, 0, stream>>>(W, scale_p, (unsigned int*)qw, W_ELEMS / 4);
    quant_x<<<M_TOT, 256, 0, stream>>>(x, qx, row_scale);

    dim3 grid(D_OUT / 128, M_TOT / 128);
    bitgemm<<<grid, 256, 0, stream>>>(qx, qw, row_scale, scale_p, b, out);
}
